// TopKRouter_86380382257744
// MI455X (gfx1250) — compile-verified, matrix operand order checked by reading, not run on hardware
//
#include <hip/hip_runtime.h>
#include <hip/hip_bf16.h>

// ---------------- problem constants ----------------
#define BATCH   4
#define SEQ     4096
#define DMODEL  4096
#define NEXP    64
#define KSEL    2
#define NTOK    (BATCH * SEQ)          // 16384
#define CAP     640                    // capacity per expert
#define NE      ((size_t)NTOK * NEXP)  // 1048576

typedef __attribute__((ext_vector_type(16))) __bf16 v16bf;
typedef __attribute__((ext_vector_type(8)))  float  v8f;

static __device__ inline unsigned fbits(float f) {
    union { float f; unsigned u; } c; c.f = f; return c.u;
}
static __device__ inline unsigned short bfbits(float f) {
    return (unsigned short)(fbits(f) >> 16);   // truncate fp32 -> bf16
}
// pack two fp32 -> dword of two bf16 (lo in [15:0], hi in [31:16])
static __device__ inline unsigned pack2bf(float lo, float hi) {
    return (fbits(hi) & 0xFFFF0000u) | (fbits(lo) >> 16);
}

struct BfPack { uint4 lo; uint4 hi; };   // 32 bytes == v16bf

static __device__ inline v16bf bf_from_pack(uint4 lo, uint4 hi) {
    BfPack s; s.lo = lo; s.hi = hi;
    v16bf r;
    __builtin_memcpy(&r, &s, sizeof(r));
    return r;
}

// ---------------- kernel 0: W fp32 -> bf16 ----------------
__global__ void convert_w(const float* __restrict__ W, unsigned short* __restrict__ Wb) {
    int i = blockIdx.x * blockDim.x + threadIdx.x;
    if (i < NEXP * DMODEL) Wb[i] = bfbits(W[i]);
}

// ---------------- kernel 1: zero kept region ----------------
__global__ void zero_f32(float* __restrict__ p, long long n) {
    long long i = (long long)blockIdx.x * blockDim.x + threadIdx.x;
    if (i < n) p[i] = 0.0f;
}

// ---------------- kernel 2: logits = x @ W^T via BF16 WMMA ----------------
// block = 128 threads (4 waves); each wave: 16 rows x 64 experts; grid = NTOK/64.
// Per 32-wide K chunk, the block stages W[64 experts x 32 K] (4 KB, bf16) in LDS
// with async global->LDS DMA (ASYNCcnt), double-buffered against the WMMAs.
// An empty asm pins all four B tiles live so the 8 ds_loads clause together and
// the 4 WMMAs issue back-to-back behind a single s_wait_dscnt.
__global__ void __launch_bounds__(128) gemm_logits(const float* __restrict__ x,
                                                   const unsigned short* __restrict__ Wb,
                                                   float* __restrict__ logits) {
    __shared__ unsigned short wlds[2][NEXP * 32];   // 2 x 4 KB

    const int tid  = threadIdx.x;
    const int lane = tid & 31;
    const int wv   = tid >> 5;                  // 0..3
    const int row0 = blockIdx.x * 64 + wv * 16; // first token row of this wave
    const int mrow = lane & 15;                 // M (or N) index within tile
    const int hsel = lane >> 4;                 // half-selector (K phase)

    v8f acc0 = {}, acc1 = {}, acc2 = {}, acc3 = {};
    const float* xrow = x + (size_t)(row0 + mrow) * DMODEL;

    // --- async DMA of one 64x32 bf16 W chunk into wlds[buf] (whole block) ---
    // 4 KB chunk = 256 x 16B pieces; 128 threads x 2 pieces each.
    auto issue_chunk = [&](int buf, int kb) {
#pragma unroll
        for (int j = 0; j < 2; ++j) {
            const int li = tid * 2 + j;            // 0..255
            const int e  = li >> 2;                // expert row (64B each)
            const int ko = (li & 3) * 8;           // k offset within chunk
            unsigned lds_off =
                (unsigned)(unsigned long long)(&wlds[buf][e * 32 + ko]);
            unsigned long long gaddr =
                (unsigned long long)(const unsigned short*)(Wb + (size_t)e * DMODEL + kb + ko);
            asm volatile("global_load_async_to_lds_b128 %0, %1, off"
                         :: "v"(lds_off), "v"(gaddr) : "memory");
        }
    };

    issue_chunk(0, 0);

    for (int kb = 0; kb < DMODEL; kb += 32) {
        const int cur = (kb >> 5) & 1;

        // own DMA for wlds[cur] retired ...
        asm volatile("s_wait_asynccnt 0" ::: "memory");
        // ... all waves' DMAs retired, all reads of wlds[cur^1] finished
        __syncthreads();
        if (kb + 32 < DMODEL) issue_chunk(cur ^ 1, kb + 32);

        // ---- A tile: 16x32 bf16. K = kb + g*16 + hsel*8 + j
        const float* pa = xrow + kb + hsel * 8;
        float4 f0 = *(const float4*)(pa + 0);
        float4 f1 = *(const float4*)(pa + 4);
        float4 f2 = *(const float4*)(pa + 16);
        float4 f3 = *(const float4*)(pa + 20);
        unsigned au[8];
        au[0] = pack2bf(f0.x, f0.y); au[1] = pack2bf(f0.z, f0.w);
        au[2] = pack2bf(f1.x, f1.y); au[3] = pack2bf(f1.z, f1.w);
        au[4] = pack2bf(f2.x, f2.y); au[5] = pack2bf(f2.z, f2.w);
        au[6] = pack2bf(f3.x, f3.y); au[7] = pack2bf(f3.z, f3.w);
        v16bf a;
        __builtin_memcpy(&a, au, sizeof(a));

        // branch-free prefetch of upcoming x (wraps within the row)
        __builtin_prefetch(xrow + ((kb + 512) & (DMODEL - 1)), 0, 1);

        // ---- load all 4 B tiles, pin them live, then 4 back-to-back WMMAs
        const unsigned short* bbase = &wlds[cur][mrow * 32 + hsel * 8];
        v16bf b0 = bf_from_pack(*(const uint4*)(bbase +    0), *(const uint4*)(bbase +   16));
        v16bf b1 = bf_from_pack(*(const uint4*)(bbase +  512), *(const uint4*)(bbase +  528));
        v16bf b2 = bf_from_pack(*(const uint4*)(bbase + 1024), *(const uint4*)(bbase + 1040));
        v16bf b3 = bf_from_pack(*(const uint4*)(bbase + 1536), *(const uint4*)(bbase + 1552));
        // force all four tiles (and A) to materialize in distinct registers
        // before any WMMA consumes them -> single s_wait_dscnt, no reg reuse
        asm volatile("" : "+v"(b0), "+v"(b1), "+v"(b2), "+v"(b3), "+v"(a));

        acc0 = __builtin_amdgcn_wmma_f32_16x16x32_bf16(false, a, false, b0, (short)0, acc0, false, false);
        acc1 = __builtin_amdgcn_wmma_f32_16x16x32_bf16(false, a, false, b1, (short)0, acc1, false, false);
        acc2 = __builtin_amdgcn_wmma_f32_16x16x32_bf16(false, a, false, b2, (short)0, acc2, false, false);
        acc3 = __builtin_amdgcn_wmma_f32_16x16x32_bf16(false, a, false, b3, (short)0, acc3, false, false);
    }

    // C/D layout: VGPR v, lanes 0-15 -> M=v, lanes 16-31 -> M=v+8 ; N = mrow
    const int rbase = row0 + hsel * 8;
#pragma unroll
    for (int v = 0; v < 8; ++v) {
        float* lr = logits + (size_t)(rbase + v) * NEXP + mrow;
        lr[0]  = acc0[v];
        lr[16] = acc1[v];
        lr[32] = acc2[v];
        lr[48] = acc3[v];
    }
}

// ---------------- kernel 3: softmax + top2 + per-expert gate sums ----------------
// 256 threads = 8 waves = 8 tokens per block; grid = NTOK/8
__global__ void softmax_top2(const float* __restrict__ logits,
                             float* __restrict__ gates,
                             unsigned long long* __restrict__ topmask,
                             float* __restrict__ gsum_partial) {
    const int lane = threadIdx.x & 31;
    const int wv   = threadIdx.x >> 5;
    const int tok  = blockIdx.x * 8 + wv;

    const float* lr = logits + (size_t)tok * NEXP;
    float l0 = lr[lane], l1 = lr[lane + 32];

    float m = fmaxf(l0, l1);
#pragma unroll
    for (int off = 16; off; off >>= 1) m = fmaxf(m, __shfl_xor(m, off, 32));

    float e0 = __expf(l0 - m), e1 = __expf(l1 - m);
    float s = e0 + e1;
#pragma unroll
    for (int off = 16; off; off >>= 1) s += __shfl_xor(s, off, 32);
    float inv = 1.0f / s;
    float g0 = e0 * inv, g1 = e1 * inv;

    gates[(size_t)tok * NEXP + lane]      = g0;
    gates[(size_t)tok * NEXP + lane + 32] = g1;

    // wave-wide top-2 with (value desc, index asc) ordering
    float a1, a2; int i1, i2;
    if (g0 >= g1) { a1 = g0; i1 = lane;      a2 = g1; i2 = lane + 32; }
    else          { a1 = g1; i1 = lane + 32; a2 = g0; i2 = lane;      }
#pragma unroll
    for (int off = 16; off; off >>= 1) {
        float b1 = __shfl_xor(a1, off, 32), b2 = __shfl_xor(a2, off, 32);
        int   j1 = __shfl_xor(i1, off, 32), j2 = __shfl_xor(i2, off, 32);
        bool agt = (a1 > b1) || (a1 == b1 && i1 < j1);
        float w1 = agt ? a1 : b1; int w1i = agt ? i1 : j1;
        float lo = agt ? b1 : a1; int loi = agt ? j1 : i1;  // losing top1
        float se = agt ? a2 : b2; int sei = agt ? i2 : j2;  // winner's second
        bool lgt = (lo > se) || (lo == se && loi < sei);
        a1 = w1; i1 = w1i;
        a2 = lgt ? lo : se; i2 = lgt ? loi : sei;
    }
    if (lane == 0) topmask[tok] = (1ull << i1) | (1ull << i2);

    // deterministic per-block per-expert gate sums
    __shared__ float gsh[8][NEXP];
    gsh[wv][lane]      = g0;
    gsh[wv][lane + 32] = g1;
    __syncthreads();
    if (threadIdx.x < NEXP) {
        float acc = 0.0f;
#pragma unroll
        for (int t = 0; t < 8; ++t) acc += gsh[t][threadIdx.x];
        gsum_partial[(size_t)blockIdx.x * NEXP + threadIdx.x] = acc;
    }
}

// ---------------- kernel 4: per-expert capacity selection ----------------
static __device__ inline unsigned cand_key(const float* gates,
                                           const unsigned long long* topmask,
                                           int t, int e) {
    if (!((topmask[t] >> e) & 1ull)) return 0u;
    return fbits(gates[(size_t)t * NEXP + e]) + 1u;   // gates > 0 -> monotone bits
}

__global__ void capacity_select(const float* __restrict__ gates,
                                const unsigned long long* __restrict__ topmask,
                                float* __restrict__ kept,
                                float* __restrict__ topidx_out,
                                float* __restrict__ valid_out,
                                int* __restrict__ cntSel) {
    const int e   = blockIdx.x;
    const int tid = threadIdx.x;            // blockDim = 256

    __shared__ unsigned hist[256];
    __shared__ unsigned prefix_s, need_s;
    __shared__ int cnt_s;
    __shared__ unsigned long long list[1024];

    if (tid == 0) { prefix_s = 0u; need_s = CAP; }
    __syncthreads();

    // 4-pass MSD radix select for CAP-th largest 32-bit key
    for (int pass = 0; pass < 4; ++pass) {
        const int shift = 24 - pass * 8;
        hist[tid] = 0u;
        __syncthreads();
        const unsigned pfx = prefix_s;
        const unsigned mhi = (pass == 0) ? 0u : (0xFFFFFFFFu << (32 - pass * 8));
        for (int t = tid; t < NTOK; t += 256) {
            unsigned key = cand_key(gates, topmask, t, e);
            if ((key & mhi) == pfx) atomicAdd(&hist[(key >> shift) & 0xFF], 1u);
        }
        __syncthreads();
        if (tid == 0) {
            unsigned need = need_s, sel = 0u;
            for (int b = 255; b >= 0; --b) {
                if (hist[b] >= need) { sel = (unsigned)b; break; }
                need -= hist[b];
            }
            prefix_s = pfx | (sel << shift);
            need_s = need;
        }
        __syncthreads();
    }
    const unsigned T = prefix_s;

    if (tid == 0) cnt_s = 0;
    __syncthreads();
    for (int t = tid; t < NTOK; t += 256) {
        unsigned key = cand_key(gates, topmask, t, e);
        if (key != 0u && key >= T) {
            int pos = atomicAdd(&cnt_s, 1);
            if (pos < 1024)
                list[pos] = ((unsigned long long)key << 32) |
                            (unsigned long long)(0xFFFFFFFFu - (unsigned)t);
        }
    }
    __syncthreads();
    const int cnt = min(cnt_s, 1024);
    for (int i = tid; i < 1024; i += 256)
        if (i >= cnt) list[i] = 0ull;
    __syncthreads();

    // bitonic sort 1024 keys, descending (deterministic: keys unique via idx)
    for (int k = 2; k <= 1024; k <<= 1) {
        for (int j = k >> 1; j > 0; j >>= 1) {
            for (int i = tid; i < 1024; i += 256) {
                int ixj = i ^ j;
                if (ixj > i) {
                    unsigned long long A = list[i], B = list[ixj];
                    bool desc = ((i & k) == 0);
                    if (desc ? (A < B) : (A > B)) { list[i] = B; list[ixj] = A; }
                }
            }
            __syncthreads();
        }
    }

    const int nvalid = min(cnt, CAP);
    if (tid == 0) cntSel[e] = nvalid;
    for (int i = tid; i < CAP; i += 256) {
        if (i < nvalid) {
            unsigned long long kk = list[i];
            int t = (int)(0xFFFFFFFFu - (unsigned)(kk & 0xFFFFFFFFull));
            topidx_out[(size_t)i * NEXP + e] = (float)t;
            valid_out[(size_t)i * NEXP + e]  = 1.0f;
            kept[(size_t)t * NEXP + e]       = 1.0f;
        } else {
            topidx_out[(size_t)i * NEXP + e] = 0.0f;
            valid_out[(size_t)i * NEXP + e]  = 0.0f;
        }
    }
}

// ---------------- kernel 5: combine = renorm(gates * kept) ----------------
__global__ void combine_renorm(const float* __restrict__ gates,
                               const float* __restrict__ kept,
                               float* __restrict__ combine) {
    const int lane = threadIdx.x & 31;
    const int wv   = threadIdx.x >> 5;
    const int tok  = blockIdx.x * 8 + wv;
    const size_t base = (size_t)tok * NEXP;

    float c0 = gates[base + lane]      * kept[base + lane];
    float c1 = gates[base + lane + 32] * kept[base + lane + 32];
    float s = c0 + c1;
#pragma unroll
    for (int off = 16; off; off >>= 1) s += __shfl_xor(s, off, 32);
    float denom = fmaxf(s, 1e-9f);
    combine[base + lane]      = c0 / denom;
    combine[base + lane + 32] = c1 / denom;
}

// ---------------- kernel 6: aux losses (fixed-order reduction) ----------------
__global__ void finalize_aux(const float* __restrict__ gsum_partial, int nparts,
                             const int* __restrict__ cntSel,
                             float* __restrict__ aux) {
    const int e = threadIdx.x;   // 64 threads
    __shared__ float ps[NEXP];
    float s = 0.0f;
    for (int i = 0; i < nparts; ++i) s += gsum_partial[(size_t)i * NEXP + e];
    float p = s / (float)NTOK;                                  // mean gate
    float af = (float)cntSel[e] / ((float)NTOK * (float)KSEL);  // assign_frac
    ps[e] = p * af;
    __syncthreads();
    if (e == 0) {
        float tot = 0.0f;
        for (int i = 0; i < NEXP; ++i) tot += ps[i];
        aux[0] = (float)NEXP * tot * 0.01f;  // aux_lb (LB_W = 0.01)
        aux[1] = 0.0f;                       // aux_z       (Z_W   = 0)
        aux[2] = 0.0f;                       // aux_entropy (ENT_W = 0)
        aux[3] = 0.0f;                       // aux_margin
    }
}

// ---------------- host launch ----------------
extern "C" void kernel_launch(void* const* d_in, const int* in_sizes, int n_in,
                              void* d_out, int out_size, void* d_ws, size_t ws_size,
                              hipStream_t stream) {
    const float* x = (const float*)d_in[0];   // [B,S,D] fp32
    const float* W = (const float*)d_in[1];   // [E,D]   fp32
    float* out = (float*)d_out;

    // output layout (reference tuple order, flattened)
    const size_t o_combine = 0;
    const size_t o_kept    = NE;
    const size_t o_aux     = 2 * NE;
    const size_t o_gates   = 2 * NE + 4;
    const size_t o_logits  = 3 * NE + 4;
    const size_t o_topidx  = 4 * NE + 4;
    const size_t o_valid   = 4 * NE + 4 + (size_t)CAP * NEXP;

    // workspace layout
    char* ws = (char*)d_ws;
    unsigned short* Wb = (unsigned short*)ws;                         // 512 KB
    unsigned long long* topmask = (unsigned long long*)(ws + 524288); // 128 KB
    float* gsum = (float*)(ws + 524288 + 131072);                     // 2048*64*4 = 512 KB
    int* cntSel = (int*)(ws + 524288 + 131072 + 524288);              // 256 B

    const int nB_softmax = NTOK / 8;   // 2048

    convert_w<<<(NEXP * DMODEL + 255) / 256, 256, 0, stream>>>(W, Wb);
    zero_f32<<<(unsigned)((NE + 255) / 256), 256, 0, stream>>>(out + o_kept, (long long)NE);
    gemm_logits<<<NTOK / 64, 128, 0, stream>>>(x, Wb, out + o_logits);
    softmax_top2<<<nB_softmax, 256, 0, stream>>>(out + o_logits, out + o_gates, topmask, gsum);
    capacity_select<<<NEXP, 256, 0, stream>>>(out + o_gates, topmask,
                                              out + o_kept, out + o_topidx, out + o_valid, cntSel);
    combine_renorm<<<NTOK / 8, 256, 0, stream>>>(out + o_gates, out + o_kept, out + o_combine);
    finalize_aux<<<1, NEXP, 0, stream>>>(gsum, nB_softmax, cntSel, out + o_aux);
}
